// ConditionalPredicateHead_47897475285053
// MI455X (gfx1250) — matrix operand order repaired and checked by reading, not compiled, not run on hardware
//
#include <hip/hip_runtime.h>

#define N_NODES 50000
#define E_EDGES 524288
#define HID 128
#define GDIM 8
#define NPRED 9
#define KDIM 264
#define W1_STRIDE 296   // bf16 elems per row of W1t (288 padded K + 8 bank-skew)
#define W2_STRIDE 136   // bf16 elems per row of W2t (128 K + 8 bank-skew)
#define H_STRIDE  136   // bf16 elems per row of staged h tile

typedef __attribute__((ext_vector_type(16))) __bf16 v16bf;
typedef __attribute__((ext_vector_type(8)))  __bf16 v8bf;
typedef __attribute__((ext_vector_type(8)))  float  v8f;
typedef __attribute__((ext_vector_type(4)))  float  v4f;
typedef __attribute__((ext_vector_type(4)))  unsigned int v4u;

union F16x16 { v16bf v; v8bf h[2]; __bf16 e[16]; };

// ---------------- prep: W1 transpose (+b1 folded at k=264) and W2 transpose ----------------
__global__ void prep_weights_kernel(const float* __restrict__ W1,   // (264,128)
                                    const float* __restrict__ b1,   // (128,)
                                    const float* __restrict__ W2,   // (128,9)
                                    __bf16* __restrict__ w1t,       // (128, W1_STRIDE)
                                    __bf16* __restrict__ w2t)       // (16, W2_STRIDE)
{
  const int stride = gridDim.x * blockDim.x;
  const int t = blockIdx.x * blockDim.x + threadIdx.x;
  for (int i = t; i < HID * W1_STRIDE; i += stride) {
    const int n = i / W1_STRIDE, k = i % W1_STRIDE;
    float v;
    if (k < KDIM)        v = W1[k * HID + n];
    else if (k == KDIM)  v = b1[n];            // bias folded: feature 264 == 1.0
    else                 v = 0.0f;
    w1t[i] = (__bf16)v;
  }
  for (int i = t; i < 16 * W2_STRIDE; i += stride) {
    const int n = i / W2_STRIDE, k = i % W2_STRIDE;
    const float v = (n < NPRED && k < HID) ? W2[k * NPRED + n] : 0.0f;
    w2t[i] = (__bf16)v;
  }
}

// ---------------- prep: node table + u -> bf16 (gathers then need no conversion) ----------
__global__ void prep_nodes_kernel(const float* __restrict__ nodes,
                                  const float* __restrict__ u,
                                  __bf16* __restrict__ nodesBf,
                                  __bf16* __restrict__ uBf)
{
  const int stride = gridDim.x * blockDim.x;
  const int t = blockIdx.x * blockDim.x + threadIdx.x;
  const int total = N_NODES * HID;
  for (int i = t; i < total; i += stride) nodesBf[i] = (__bf16)nodes[i];
  for (int i = t; i < 64 * GDIM; i += stride) uBf[i] = (__bf16)u[i];
}

// 8 features at offset o of the virtual 288-wide row (f32 source, converts).
__device__ __forceinline__ void load_chunk8_f32(const float* __restrict__ srcRow,
                                                const float* __restrict__ tgtRow,
                                                const float* __restrict__ uRow,
                                                int o, __bf16* dst)
{
  const float* p;
  if (o < 128)        p = srcRow + o;
  else if (o < 256)   p = tgtRow + (o - 128);
  else if (o == 256)  p = uRow;
  else {
#pragma unroll
    for (int i = 0; i < 8; ++i) dst[i] = (__bf16)0.0f;
    if (o == KDIM) dst[0] = (__bf16)1.0f;      // bias feature
    return;
  }
  const v4f x = *(const v4f*)p;
  const v4f y = *(const v4f*)(p + 4);
  dst[0] = (__bf16)x.x; dst[1] = (__bf16)x.y; dst[2] = (__bf16)x.z; dst[3] = (__bf16)x.w;
  dst[4] = (__bf16)y.x; dst[5] = (__bf16)y.y; dst[6] = (__bf16)y.z; dst[7] = (__bf16)y.w;
}

// 8 features at offset o (bf16 source, single 16B load, no conversion).
__device__ __forceinline__ v8bf load_chunk8_bf(const __bf16* __restrict__ srcRow,
                                               const __bf16* __restrict__ tgtRow,
                                               const __bf16* __restrict__ uRow,
                                               int o)
{
  if (o < 128)  return *(const v8bf*)(srcRow + o);
  if (o < 256)  return *(const v8bf*)(tgtRow + (o - 128));
  if (o == 256) return *(const v8bf*)uRow;
  v8bf z;
#pragma unroll
  for (int i = 0; i < 8; ++i) z[i] = (__bf16)0.0f;
  if (o == KDIM) z[0] = (__bf16)1.0f;          // bias feature
  return z;
}

__device__ __forceinline__ F16x16 load_bfrag(const __bf16* __restrict__ base,
                                             int row, int rstride, int koff)
{
  F16x16 b;
  const __bf16* bp = base + row * rstride + koff;
  b.h[0] = *(const v8bf*)bp;
  b.h[1] = *(const v8bf*)(bp + 16);
  return b;
}

// ---------------- main fused edge-MLP kernel ----------------
template <bool BFG>
__global__ void __launch_bounds__(256, 1) edge_mlp_kernel(
    const float*  __restrict__ nodes,    // (N,128) f32
    const __bf16* __restrict__ nodesBf,  // (N,128) bf16 (valid iff BFG)
    const int*    __restrict__ eidx,     // (2,E)
    const float*  __restrict__ u,        // (64,8) f32
    const __bf16* __restrict__ uBf,      // (64,8) bf16 (valid iff BFG)
    const int*    __restrict__ batch,    // (N,)
    const float*  __restrict__ b2,       // (9,)
    const __bf16* __restrict__ w1t,      // (128, W1_STRIDE)
    const __bf16* __restrict__ w2t,      // (16, W2_STRIDE)
    float* __restrict__ out)             // (E,9)
{
  extern __shared__ __align__(16) char smem[];
  __bf16* sW1 = (__bf16*)smem;                 // 128*296 bf16
  __bf16* sW2 = sW1 + HID * W1_STRIDE;         // 16*136 bf16
  __bf16* sH  = sW2 + 16 * W2_STRIDE;          // 8 waves * 32 * 136 bf16

  {
    const int nv1 = (HID * W1_STRIDE) / 8;
    const v4u* g1 = (const v4u*)w1t;
    v4u* l1 = (v4u*)sW1;
    for (int i = threadIdx.x; i < nv1; i += blockDim.x) l1[i] = g1[i];
    const int nv2 = (16 * W2_STRIDE) / 8;
    const v4u* g2 = (const v4u*)w2t;
    v4u* l2 = (v4u*)sW2;
    for (int i = threadIdx.x; i < nv2; i += blockDim.x) l2[i] = g2[i];
  }
  __syncthreads();

  const int lane   = threadIdx.x & 31;
  const int waveId = threadIdx.x >> 5;
  const int r      = lane & 15;
  const bool hiHalf = (lane >= 16);
  const int bko = hiHalf ? 8 : 0;              // B-frag K sub-offset
  __bf16* hW = sH + waveId * 32 * H_STRIDE;

  const float b2r = (r < NPRED) ? b2[r] : 0.0f;
  const v8f vzero = {0.f, 0.f, 0.f, 0.f, 0.f, 0.f, 0.f, 0.f};

  const int numWaves = gridDim.x * (blockDim.x >> 5);
  const int gw = blockIdx.x * (blockDim.x >> 5) + waveId;
  const int numPairs = E_EDGES / 32;

  for (int p = gw; p < numPairs; p += numWaves) {
    const int base = p * 32;
    const int e0 = base + r, e1 = base + 16 + r;
    const int s0 = eidx[e0], t0 = eidx[E_EDGES + e0];
    const int s1 = eidx[e1], t1 = eidx[E_EDGES + e1];
    const float*  src0f = nodes + (size_t)s0 * HID;
    const float*  tgt0f = nodes + (size_t)t0 * HID;
    const float*  u0f   = u + (size_t)batch[s0] * GDIM;
    const float*  src1f = nodes + (size_t)s1 * HID;
    const float*  tgt1f = nodes + (size_t)t1 * HID;
    const float*  u1f   = u + (size_t)batch[s1] * GDIM;
    const __bf16* src0b = nodesBf + (size_t)s0 * HID;
    const __bf16* tgt0b = nodesBf + (size_t)t0 * HID;
    const __bf16* u0b   = uBf + (size_t)batch[s0] * GDIM;
    const __bf16* src1b = nodesBf + (size_t)s1 * HID;
    const __bf16* tgt1b = nodesBf + (size_t)t1 * HID;
    const __bf16* u1b   = uBf + (size_t)batch[s1] * GDIM;

    v8f acc0[8], acc1[8];
#pragma unroll
    for (int nb = 0; nb < 8; ++nb) { acc0[nb] = vzero; acc1[nb] = vzero; }

    // A-fragment builder for K-block kb (both M-tiles)
    auto buildA = [&](int kb, F16x16& a0, F16x16& a1) {
      const int o1 = kb * 32 + (hiHalf ? 8 : 0);
      const int o2 = o1 + 16;
      if constexpr (BFG) {
        a0.h[0] = load_chunk8_bf(src0b, tgt0b, u0b, o1);
        a0.h[1] = load_chunk8_bf(src0b, tgt0b, u0b, o2);
        a1.h[0] = load_chunk8_bf(src1b, tgt1b, u1b, o1);
        a1.h[1] = load_chunk8_bf(src1b, tgt1b, u1b, o2);
      } else {
        load_chunk8_f32(src0f, tgt0f, u0f, o1, &a0.e[0]);
        load_chunk8_f32(src0f, tgt0f, u0f, o2, &a0.e[8]);
        load_chunk8_f32(src1f, tgt1f, u1f, o1, &a1.e[0]);
        load_chunk8_f32(src1f, tgt1f, u1f, o2, &a1.e[8]);
      }
    };

    // -------- layer 1: (32 x 288) @ (288 x 128), double-buffered B frags --------
    F16x16 a0, a1, a0n, a1n, bcur, bnext;
    buildA(0, a0, a1);
    bcur = load_bfrag(sW1, r, W1_STRIDE, bko);          // (kb=0, nb=0)
#pragma unroll
    for (int kb = 0; kb < 9; ++kb) {
      if (kb + 1 < 9) buildA(kb + 1, a0n, a1n);         // prefetch next A (global)
#pragma unroll
      for (int nb = 0; nb < 8; ++nb) {
        const int nkb = (nb + 1 < 8) ? kb : ((kb + 1 < 9) ? kb + 1 : 0);
        const int nnb = (nb + 1 < 8) ? nb + 1 : 0;
        bnext = load_bfrag(sW1, nnb * 16 + r, W1_STRIDE, nkb * 32 + bko);
        acc0[nb] = __builtin_amdgcn_wmma_f32_16x16x32_bf16(
            false, a0.v, false, bcur.v, (short)0, acc0[nb], false, false);
        acc1[nb] = __builtin_amdgcn_wmma_f32_16x16x32_bf16(
            false, a1.v, false, bcur.v, (short)0, acc1[nb], false, false);
        bcur = bnext;
      }
      a0 = a0n; a1 = a1n;
    }

    // -------- preload layer-2 B frags (shared by both M-tiles) --------
    F16x16 bl2[4];
#pragma unroll
    for (int kb = 0; kb < 4; ++kb)
      bl2[kb] = load_bfrag(sW2, r, W2_STRIDE, kb * 32 + bko);

    // -------- ReLU + stage h to per-wave LDS (bf16); bias already in GEMM ------
#pragma unroll
    for (int nb = 0; nb < 8; ++nb) {
      const int n = nb * 16 + r;
#pragma unroll
      for (int j = 0; j < 8; ++j) {
        const int m = j + (hiHalf ? 8 : 0);
        float v0 = acc0[nb][j]; v0 = v0 > 0.f ? v0 : 0.f;
        float v1 = acc1[nb][j]; v1 = v1 > 0.f ? v1 : 0.f;
        hW[m * H_STRIDE + n]        = (__bf16)v0;
        hW[(16 + m) * H_STRIDE + n] = (__bf16)v1;
      }
    }

    // -------- layer 2: (32 x 128) @ (128 x 16[9]) --------
#pragma unroll
    for (int t = 0; t < 2; ++t) {
      F16x16 at[4];
#pragma unroll
      for (int kb = 0; kb < 4; ++kb) {
        const int k1 = kb * 32 + (hiHalf ? 8 : 0);
        const __bf16* ap = hW + (t * 16 + r) * H_STRIDE + k1;
        at[kb].h[0] = *(const v8bf*)ap;
        at[kb].h[1] = *(const v8bf*)(ap + 16);
      }
      v8f o = vzero;
#pragma unroll
      for (int kb = 0; kb < 4; ++kb)
        o = __builtin_amdgcn_wmma_f32_16x16x32_bf16(
            false, at[kb].v, false, bl2[kb].v, (short)0, o, false, false);
      if (r < NPRED) {
#pragma unroll
        for (int j = 0; j < 8; ++j) {
          const int m = j + (hiHalf ? 8 : 0);
          out[(size_t)(base + t * 16 + m) * NPRED + r] = o[j] + b2r;
        }
      }
    }
  }
}

extern "C" void kernel_launch(void* const* d_in, const int* in_sizes, int n_in,
                              void* d_out, int out_size, void* d_ws, size_t ws_size,
                              hipStream_t stream) {
  const float* nodes = (const float*)d_in[0];
  const int*   eidx  = (const int*)d_in[1];
  const float* u     = (const float*)d_in[2];
  const int*   batch = (const int*)d_in[3];
  const float* W1    = (const float*)d_in[4];
  const float* b1    = (const float*)d_in[5];
  const float* W2    = (const float*)d_in[6];
  const float* b2    = (const float*)d_in[7];
  float* out = (float*)d_out;

  __bf16* w1t = (__bf16*)d_ws;                       // 75776 B
  __bf16* w2t = w1t + HID * W1_STRIDE;               // +4352 B
  __bf16* nodesBf = w2t + 16 * W2_STRIDE;            // +12.8 MB
  __bf16* uBf = nodesBf + (size_t)N_NODES * HID;     // +1 KB

  const size_t needBf = (size_t)(HID * W1_STRIDE + 16 * W2_STRIDE +
                                 N_NODES * HID + 64 * GDIM) * sizeof(__bf16);
  const bool useBf = (ws_size >= needBf);

  prep_weights_kernel<<<64, 256, 0, stream>>>(W1, b1, W2, w1t, w2t);

  const size_t smemB =
      (size_t)(HID * W1_STRIDE + 16 * W2_STRIDE + 8 * 32 * H_STRIDE) * sizeof(__bf16);

  if (useBf) {
    prep_nodes_kernel<<<1024, 256, 0, stream>>>(nodes, u, nodesBf, uBf);
    hipFuncSetAttribute(reinterpret_cast<const void*>(&edge_mlp_kernel<true>),
                        hipFuncAttributeMaxDynamicSharedMemorySize, (int)smemB);
    edge_mlp_kernel<true><<<1024, 256, smemB, stream>>>(
        nodes, nodesBf, eidx, u, uBf, batch, b2, w1t, w2t, out);
  } else {
    hipFuncSetAttribute(reinterpret_cast<const void*>(&edge_mlp_kernel<false>),
                        hipFuncAttributeMaxDynamicSharedMemorySize, (int)smemB);
    edge_mlp_kernel<false><<<1024, 256, smemB, stream>>>(
        nodes, nodesBf, eidx, u, uBf, batch, b2, w1t, w2t, out);
  }
}
